// Encoder_21371757265491
// MI455X (gfx1250) — matrix operand order repaired and checked by reading, not compiled, not run on hardware
//
#include <hip/hip_runtime.h>
#include <hip/hip_bf16.h>
#include <stdint.h>
#include <stddef.h>

// ---------------------------------------------------------------------------
// CDNA5 (gfx1250) types for WMMA
// ---------------------------------------------------------------------------
typedef __bf16 bf16_t;
typedef __attribute__((ext_vector_type(16))) __bf16 v16bf;
typedef __attribute__((ext_vector_type(8)))  __bf16 v8bf;
typedef __attribute__((ext_vector_type(8)))  float  v8f;

#define B_    256
#define TM1   256
#define D_    128
#define H_    128
#define M_TOT (B_ * TM1)   // 65536 rows for the batched-token GEMMs

#define V8F_ZERO {0.f,0.f,0.f,0.f,0.f,0.f,0.f,0.f}

// D = A(16x32 bf16) x B(32x16 bf16) + C(16x16 f32)
__device__ __forceinline__ v8f wmma_bf16(v16bf a, v16bf b, v8f c) {
  return __builtin_amdgcn_wmma_f32_16x16x32_bf16(false, a, false, b, (short)0, c,
                                                 false, false);
}

// A fragment (16x32, row-major source [M,K]): lane L (m = L%16)
// holds K = [c*16, c*16+16) with c = L/16 (same order as the B fragment).
// Loaded as two 16-byte contiguous chunks.
__device__ __forceinline__ v16bf load_a_frag(const bf16_t* __restrict__ base,
                                             int ld, int m0, int k0) {
  const int lane = threadIdx.x & 31;
  const int half = lane >> 4;
  const bf16_t* p = base + (size_t)(m0 + (lane & 15)) * ld + k0 + half * 16;
  union { v16bf v; v8bf h[2]; } f;
  f.h[0] = *(const v8bf*)(p);
  f.h[1] = *(const v8bf*)(p + 8);
  return f.v;
}

// B fragment (32x16): lane L holds column n = L%16, K = [(L/16)*16, +16).
// With weights stored [N,K] row-major this is one contiguous 32-byte load
// (computes A x W^T, which is exactly x @ W.T in the reference).
__device__ __forceinline__ v16bf load_b_frag(const bf16_t* __restrict__ W,
                                             int ld, int n0, int k0) {
  const int lane = threadIdx.x & 31;
  const bf16_t* p = W + (size_t)(n0 + (lane & 15)) * ld + k0 + (lane >> 4) * 16;
  return *(const v16bf*)(p);
}

// Fast activations. gfx1250 has a hardware V_TANH_F32 (TRANS unit); ride it
// for both tanh and sigmoid (sigmoid(x) = 0.5*(1+tanh(x/2))) when the builtin
// exists, else fall back to v_exp_f32.
__device__ __forceinline__ float tanh_f(float x) {
#if __has_builtin(__builtin_amdgcn_tanhf)
  return __builtin_amdgcn_tanhf(x);
#else
  x = fminf(fmaxf(x, -15.f), 15.f);
  float e = __expf(2.f * x);
  return (e - 1.f) / (e + 1.f);
#endif
}
__device__ __forceinline__ float sigmoid_f(float x) {
#if __has_builtin(__builtin_amdgcn_tanhf)
  return 0.5f * (1.0f + __builtin_amdgcn_tanhf(0.5f * x));
#else
  x = fminf(fmaxf(x, -30.f), 30.f);
  return 1.f / (1.f + __expf(-x));
#endif
}

// CDNA5 async global->LDS copy (ASYNCcnt protocol, CDNA5 ISA).
// VDST vgpr carries the LDS byte address, VADDR the 64-bit global address.
__device__ __forceinline__ void async_load_b128_to_lds(uint32_t lds_addr,
                                                       const void* gaddr) {
  asm volatile("global_load_async_to_lds_b128 %0, %1, off"
               :
               : "v"(lds_addr), "v"((uint64_t)(uintptr_t)gaddr)
               : "memory");
}
__device__ __forceinline__ void wait_asynccnt0() {
  asm volatile("s_wait_asynccnt 0x0" ::: "memory");
}

// ---------------------------------------------------------------------------
// Prep: X -> bf16 (row-major and d-major transpose), weights -> bf16, biases
// ---------------------------------------------------------------------------
__global__ void convert_X_kernel(const float* __restrict__ X,
                                 bf16_t* __restrict__ Xbf,
                                 bf16_t* __restrict__ XbfT) {
  const size_t N = (size_t)B_ * TM1 * D_;
  for (size_t i = (size_t)blockIdx.x * blockDim.x + threadIdx.x; i < N;
       i += (size_t)gridDim.x * blockDim.x) {
    float v = X[i];
    bf16_t bv = (bf16_t)v;
    Xbf[i] = bv;
    int d = (int)(i & 127);
    int t = (int)((i >> 7) & 255);
    int b = (int)(i >> 15);
    XbfT[(((size_t)b * D_ + d) << 8) | t] = bv;  // [B, D, Tm1]
  }
}

__global__ void prep_weights_kernel(
    const float* W_ih1, const float* b_ih1, const float* b_hh1,
    const float* W_we, const float* W_ue,
    const float* W_ih2, const float* b_ih2,
    const float* W_hh2, const float* b_hh2,
    bf16_t* W1bf, bf16_t* Wwebf, bf16_t* Wuebf, bf16_t* Wih2bf, bf16_t* Whh2bf,
    float* bias1, float* bias2, bf16_t* h0, float* c0) {
  const int idx = blockIdx.x * blockDim.x + threadIdx.x;  // grid covers 65536
  if (idx < 4 * H_ * D_) {          // all five weight mats are 65536 elements
    W1bf[idx]   = (bf16_t)W_ih1[idx];
    Wwebf[idx]  = (bf16_t)W_we[idx];
    Wuebf[idx]  = (bf16_t)W_ue[idx];
    Wih2bf[idx] = (bf16_t)W_ih2[idx];
    Whh2bf[idx] = (bf16_t)W_hh2[idx];
  }
  if (idx < 4 * H_) {
    bias1[idx] = b_ih1[idx] + b_hh1[idx];
    bias2[idx] = b_ih2[idx] + b_hh2[idx];
  }
  if (idx < TM1 * H_) {             // zero initial LSTM state
    h0[idx] = (bf16_t)0.0f;
    c0[idx] = 0.0f;
  }
}

// ---------------------------------------------------------------------------
// Phase 1: encoder gates GEMM [65536x128]x[128x512] fused with cell math.
// c_prev == 0, so the f gate is dead: only i, g, o tiles are accumulated.
// Wave w owns column tile c0 = w*16 and accumulates the 3 gate tiles at
// n = {0,256,384}+c0 so the activation fuses entirely in registers.
// Writes hc = [h_s | c_s] bf16 [65536, 256] for the attention GEMM.
// ---------------------------------------------------------------------------
__global__ void encoder_gates_kernel(const bf16_t* __restrict__ Xbf,
                                     const bf16_t* __restrict__ W1,
                                     const float* __restrict__ bias1,
                                     bf16_t* __restrict__ hc) {
  const int wave = threadIdx.x >> 5;   // 0..7 -> column tile
  const int lane = threadIdx.x & 31;
  const int m0 = blockIdx.x * 16;
  const int c0 = wave * 16;
  const int gate_n[3] = {0 * H_, 2 * H_, 3 * H_};  // i, g, o rows of W_ih1

  // Warm the (L2-resident, reused-by-all-waves) weight rows.
  __builtin_prefetch(W1 + (size_t)c0 * D_, 0, 3);
  __builtin_prefetch(W1 + (size_t)(2 * H_ + c0) * D_, 0, 3);
  __builtin_prefetch(W1 + (size_t)(3 * H_ + c0) * D_, 0, 3);

  v8f acc[3] = {V8F_ZERO, V8F_ZERO, V8F_ZERO};
#pragma unroll
  for (int kt = 0; kt < 4; ++kt) {
    const int k0 = kt * 32;
    v16bf a = load_a_frag(Xbf, D_, m0, k0);
#pragma unroll
    for (int g = 0; g < 3; ++g) {
      v16bf b = load_b_frag(W1, D_, gate_n[g] + c0, k0);
      acc[g] = wmma_bf16(a, b, acc[g]);
    }
  }
  const int col = c0 + (lane & 15);
  const int rbase = (lane >> 4) * 8;
  const float bi = bias1[0 * H_ + col];
  const float bg = bias1[2 * H_ + col];
  const float bo = bias1[3 * H_ + col];
#pragma unroll
  for (int r = 0; r < 8; ++r) {
    const int m = m0 + rbase + r;
    float cs = sigmoid_f(acc[0][r] + bi) * tanh_f(acc[1][r] + bg);
    float hs = sigmoid_f(acc[2][r] + bo) * tanh_f(cs);
    bf16_t* row = hc + (size_t)m * (2 * H_);
    row[col] = (bf16_t)hs;
    row[H_ + col] = (bf16_t)cs;
  }
}

// ---------------------------------------------------------------------------
// Phase 2a: A = hc @ W_we^T + b_we   -> bf16 [65536, 256]
// ---------------------------------------------------------------------------
__global__ void attn_A_kernel(const bf16_t* __restrict__ hc,
                              const bf16_t* __restrict__ Wwe,
                              const float* __restrict__ b_we,
                              bf16_t* __restrict__ Abf) {
  const int wave = threadIdx.x >> 5;   // 0..15 -> s tile
  const int lane = threadIdx.x & 31;
  const int m0 = blockIdx.x * 16;
  const int s0 = wave * 16;
  __builtin_prefetch(Wwe + (size_t)s0 * (2 * H_), 0, 3);
  v8f acc = V8F_ZERO;
#pragma unroll
  for (int kt = 0; kt < 8; ++kt) {
    v16bf a = load_a_frag(hc, 2 * H_, m0, kt * 32);
    v16bf b = load_b_frag(Wwe, 2 * H_, s0, kt * 32);
    acc = wmma_bf16(a, b, acc);
  }
  const int s = s0 + (lane & 15);
  const int rbase = (lane >> 4) * 8;
  const float bw = b_we[s];
#pragma unroll
  for (int r = 0; r < 8; ++r)
    Abf[(size_t)(m0 + rbase + r) * TM1 + s] = (bf16_t)(acc[r] + bw);
}

// ---------------------------------------------------------------------------
// Phase 2b: U[b] = X[b]^T @ W_ue^T + b_ue  -> bf16 [256][128, 256]
// A operand is the pre-transposed XbfT (K = t contiguous).
// ---------------------------------------------------------------------------
__global__ void attn_U_kernel(const bf16_t* __restrict__ XbfT,
                              const bf16_t* __restrict__ Wue,
                              const float* __restrict__ b_ue,
                              bf16_t* __restrict__ Ubf) {
  const int b = blockIdx.x;
  const int m0 = blockIdx.y * 16;       // d tile
  const int wave = threadIdx.x >> 5;    // 0..15 -> s tile
  const int lane = threadIdx.x & 31;
  const int s0 = wave * 16;
  __builtin_prefetch(Wue + (size_t)s0 * TM1, 0, 3);
  const bf16_t* Ab = XbfT + (size_t)b * D_ * TM1;
  v8f acc = V8F_ZERO;
#pragma unroll
  for (int kt = 0; kt < 8; ++kt) {
    v16bf a = load_a_frag(Ab, TM1, m0, kt * 32);
    v16bf bb = load_b_frag(Wue, TM1, s0, kt * 32);
    acc = wmma_bf16(a, bb, acc);
  }
  const int s = s0 + (lane & 15);
  const int rbase = (lane >> 4) * 8;
  const float bu = b_ue[s];
  bf16_t* Ub = Ubf + (size_t)b * D_ * TM1;
#pragma unroll
  for (int r = 0; r < 8; ++r)
    Ub[(size_t)(m0 + rbase + r) * TM1 + s] = (bf16_t)(acc[r] + bu);
}

// ---------------------------------------------------------------------------
// Phase 3: attention steps. scan carry is None => all (b,t) independent.
// One block per (b,t); thread d computes score[d] = sum_s v[s]*tanh(a[s]+U[d,s]),
// then block-wide softmax over d, X_tilde = w * x_t  (bf16 for the final GEMM).
// The per-block A row (512B bf16) and v vector (1KB f32) are staged into LDS
// with CDNA5 async global->LDS copies (ASYNCcnt + barrier protocol).
// ---------------------------------------------------------------------------
__global__ void attn_softmax_kernel(const float* __restrict__ X,
                                    const bf16_t* __restrict__ Abf,
                                    const bf16_t* __restrict__ Ubf,
                                    const float* __restrict__ Wve,
                                    const float* __restrict__ b_ve,
                                    bf16_t* __restrict__ Xt) {
  __shared__ __align__(16) bf16_t a_raw[TM1];  // staged A row (bf16)
  __shared__ float a_s[TM1];                   // converted to f32
  __shared__ __align__(16) float v_s[TM1];     // staged v (f32)
  __shared__ float red[D_];
  const int bt = blockIdx.x;       // b*256 + t
  const int b = bt >> 8;
  const int d = threadIdx.x;       // 0..127
  const int lane = threadIdx.x & 31;

  const bf16_t* arow = Abf + (size_t)bt * TM1;
  // Wave 0 async-stages the 512B A row; wave 1 async-stages the 1KB v vector.
  if (threadIdx.x < 32) {
    async_load_b128_to_lds((uint32_t)(uintptr_t)&a_raw[lane * 8],
                           arow + lane * 8);
  } else if (threadIdx.x < 64) {
    async_load_b128_to_lds((uint32_t)(uintptr_t)&v_s[lane * 4],
                           Wve + lane * 4);
    async_load_b128_to_lds((uint32_t)(uintptr_t)&v_s[128 + lane * 4],
                           Wve + 128 + lane * 4);
  }
  wait_asynccnt0();          // issuing waves drain ASYNCcnt before the barrier
  __syncthreads();
  a_s[d] = (float)a_raw[d];
  a_s[d + D_] = (float)a_raw[d + D_];
  __syncthreads();

  const v8bf* up = (const v8bf*)(Ubf + ((size_t)b * D_ + d) * TM1);
  float acc = 0.f;
#pragma unroll 4
  for (int s8 = 0; s8 < TM1 / 8; ++s8) {
    v8bf u = up[s8];
    const int sb = s8 * 8;
#pragma unroll
    for (int j = 0; j < 8; ++j)
      acc += v_s[sb + j] * tanh_f(a_s[sb + j] + (float)u[j]);
  }
  float score = acc + b_ve[0];

  // softmax over the 128 d-lanes (4 waves) via LDS tree reduction
  red[d] = score;
  __syncthreads();
  for (int off = D_ / 2; off > 0; off >>= 1) {
    if (d < off) red[d] = fmaxf(red[d], red[d + off]);
    __syncthreads();
  }
  const float mx = red[0];
  __syncthreads();
  const float e = __expf(score - mx);
  red[d] = e;
  __syncthreads();
  for (int off = D_ / 2; off > 0; off >>= 1) {
    if (d < off) red[d] += red[d + off];
    __syncthreads();
  }
  const float w = e / red[0];
  const size_t xi = (size_t)bt * D_ + d;
  Xt[xi] = (bf16_t)(w * X[xi]);
}

// ---------------------------------------------------------------------------
// Phase 4: final LSTM step (sequential over b; one launch per step).
// gates = Xt[b] @ W_ih2^T + h @ W_hh2^T + bias, fused cell update.
// Wave owns column tile c0; accumulates all four gate tiles -> register fuse.
// ---------------------------------------------------------------------------
__global__ void lstm_step_kernel(const bf16_t* __restrict__ Xt_b,
                                 const bf16_t* __restrict__ Wih2,
                                 const bf16_t* __restrict__ Whh2,
                                 const float* __restrict__ bias2,
                                 const bf16_t* __restrict__ h_in,
                                 float* __restrict__ c_state,
                                 bf16_t* __restrict__ h_out,
                                 float* __restrict__ out_b) {
  const int wave = threadIdx.x >> 5;   // 0..7 -> column tile
  const int lane = threadIdx.x & 31;
  const int m0 = blockIdx.x * 16;      // rows over Tm1 (the scan's "batch")
  const int c0 = wave * 16;
  __builtin_prefetch(Wih2 + (size_t)c0 * D_, 0, 3);
  __builtin_prefetch(Whh2 + (size_t)c0 * H_, 0, 3);

  v8f acc[4] = {V8F_ZERO, V8F_ZERO, V8F_ZERO, V8F_ZERO};
#pragma unroll
  for (int kt = 0; kt < 4; ++kt) {     // x path
    v16bf a = load_a_frag(Xt_b, D_, m0, kt * 32);
#pragma unroll
    for (int g = 0; g < 4; ++g) {
      v16bf b = load_b_frag(Wih2, D_, g * H_ + c0, kt * 32);
      acc[g] = wmma_bf16(a, b, acc[g]);
    }
  }
#pragma unroll
  for (int kt = 0; kt < 4; ++kt) {     // h path
    v16bf a = load_a_frag(h_in, H_, m0, kt * 32);
#pragma unroll
    for (int g = 0; g < 4; ++g) {
      v16bf b = load_b_frag(Whh2, H_, g * H_ + c0, kt * 32);
      acc[g] = wmma_bf16(a, b, acc[g]);
    }
  }
  const int col = c0 + (lane & 15);
  const int rbase = (lane >> 4) * 8;
  const float bi = bias2[0 * H_ + col];
  const float bff = bias2[1 * H_ + col];
  const float bg = bias2[2 * H_ + col];
  const float bo = bias2[3 * H_ + col];
#pragma unroll
  for (int r = 0; r < 8; ++r) {
    const int s = m0 + rbase + r;
    const size_t idx = (size_t)s * H_ + col;
    const float cold = c_state[idx];
    const float cn =
        sigmoid_f(acc[1][r] + bff) * cold + sigmoid_f(acc[0][r] + bi) * tanh_f(acc[2][r] + bg);
    const float hn = sigmoid_f(acc[3][r] + bo) * tanh_f(cn);
    c_state[idx] = cn;
    h_out[idx] = (bf16_t)hn;
    out_b[idx] = hn;   // enc_h[b, s, col]
  }
}

// ---------------------------------------------------------------------------
// Host launcher
// ---------------------------------------------------------------------------
extern "C" void kernel_launch(void* const* d_in, const int* in_sizes, int n_in,
                              void* d_out, int out_size, void* d_ws, size_t ws_size,
                              hipStream_t stream) {
  const float* X     = (const float*)d_in[0];
  const float* W_ih1 = (const float*)d_in[1];
  const float* b_ih1 = (const float*)d_in[2];
  // d_in[3] = W_hh1 : unused (zero encoder state)
  const float* b_hh1 = (const float*)d_in[4];
  const float* W_we  = (const float*)d_in[5];
  const float* b_we  = (const float*)d_in[6];
  const float* W_ue  = (const float*)d_in[7];
  const float* b_ue  = (const float*)d_in[8];
  const float* W_ve  = (const float*)d_in[9];
  const float* b_ve  = (const float*)d_in[10];
  const float* W_ih2 = (const float*)d_in[11];
  const float* b_ih2 = (const float*)d_in[12];
  const float* W_hh2 = (const float*)d_in[13];
  const float* b_hh2 = (const float*)d_in[14];

  char* ws = (char*)d_ws;
  size_t off = 0;
  auto carve = [&](size_t bytes) -> void* {
    void* p = ws + off;
    off = (off + bytes + 255) & ~(size_t)255;
    return p;
  };
  bf16_t* Xbf    = (bf16_t*)carve((size_t)M_TOT * D_ * 2);        // [65536,128]
  bf16_t* XbfT   = (bf16_t*)carve((size_t)B_ * D_ * TM1 * 2);     // [256,128,256]
  bf16_t* hc     = (bf16_t*)carve((size_t)M_TOT * 2 * H_ * 2);    // [65536,256]
  bf16_t* Abf    = (bf16_t*)carve((size_t)M_TOT * TM1 * 2);       // [65536,256]
  bf16_t* Ubf    = (bf16_t*)carve((size_t)B_ * D_ * TM1 * 2);     // [256,128,256]
  bf16_t* Xt     = (bf16_t*)carve((size_t)M_TOT * D_ * 2);        // [65536,128]
  bf16_t* W1bf   = (bf16_t*)carve((size_t)4 * H_ * D_ * 2);
  bf16_t* Wwebf  = (bf16_t*)carve((size_t)TM1 * 2 * H_ * 2);
  bf16_t* Wuebf  = (bf16_t*)carve((size_t)TM1 * TM1 * 2);
  bf16_t* Wih2bf = (bf16_t*)carve((size_t)4 * H_ * D_ * 2);
  bf16_t* Whh2bf = (bf16_t*)carve((size_t)4 * H_ * H_ * 2);
  float*  bias1  = (float*)carve((size_t)4 * H_ * 4);
  float*  bias2  = (float*)carve((size_t)4 * H_ * 4);
  bf16_t* hbuf0  = (bf16_t*)carve((size_t)TM1 * H_ * 2);
  bf16_t* hbuf1  = (bf16_t*)carve((size_t)TM1 * H_ * 2);
  float*  cstate = (float*)carve((size_t)TM1 * H_ * 4);
  (void)ws_size; (void)n_in; (void)in_sizes; (void)out_size;

  // Prep: bf16 conversions, transposed X copy, fused biases, zeroed state.
  convert_X_kernel<<<4096, 256, 0, stream>>>(X, Xbf, XbfT);
  prep_weights_kernel<<<256, 256, 0, stream>>>(
      W_ih1, b_ih1, b_hh1, W_we, W_ue, W_ih2, b_ih2, W_hh2, b_hh2,
      W1bf, Wwebf, Wuebf, Wih2bf, Whh2bf, bias1, bias2, hbuf0, cstate);

  // Phase 1: encoder gates + cell fuse (f gate eliminated; c_prev == 0).
  encoder_gates_kernel<<<M_TOT / 16, 256, 0, stream>>>(Xbf, W1bf, bias1, hc);

  // Phase 2: attention precompute GEMMs.
  attn_A_kernel<<<M_TOT / 16, 512, 0, stream>>>(hc, Wwebf, b_we, Abf);
  attn_U_kernel<<<dim3(B_, D_ / 16), 512, 0, stream>>>(XbfT, Wuebf, b_ue, Ubf);

  // Phase 3: all (b,t) attention steps in parallel.
  attn_softmax_kernel<<<M_TOT, D_, 0, stream>>>(X, Abf, Ubf, W_ve, b_ve, Xt);

  // Phase 4: final LSTM, sequential over B with ping-pong h state.
  for (int b = 0; b < B_; ++b) {
    const bf16_t* hin = (b & 1) ? hbuf1 : hbuf0;
    bf16_t* hout = (b & 1) ? hbuf0 : hbuf1;
    lstm_step_kernel<<<TM1 / 16, 256, 0, stream>>>(
        Xt + (size_t)b * TM1 * D_, Wih2bf, Whh2bf, bias2, hin, cstate, hout,
        (float*)d_out + (size_t)b * TM1 * H_);
  }
}
